// M2MRF_Module_21139829031564
// MI455X (gfx1250) — compile-verified, matrix operand order checked by reading, not run on hardware
//
#include <hip/hip_runtime.h>
#include <stdint.h>

// ---------------------------------------------------------------------------
// M2MRF fused kernel for gfx1250 (MI455X):
//   unfold(4x4) -> GEMM1 (1024x1024) -> GEMM2 (256x1024) -> fold(2x2)
// bf16 WMMA (v_wmma_f32_16x16x32_bf16), f32 accumulate, y kept in LDS.
// ---------------------------------------------------------------------------

typedef __bf16 v16bf __attribute__((ext_vector_type(16)));
typedef float  v8f   __attribute__((ext_vector_type(8)));

#define C_IN   64
#define Hh     512
#define Ww     512
#define K1     1024   // cin  = 16*C
#define FC     1024   // hidden channels
#define CO     256    // cout
#define LT     64     // patch columns per block
#define HO     256
#define WO     256

__device__ __forceinline__ unsigned short f2bf(float f) {
    unsigned int u = __float_as_uint(f);
    u += 0x7FFFu + ((u >> 16) & 1u);      // round-to-nearest-even
    return (unsigned short)(u >> 16);
}

// --- weight f32 -> bf16 pre-conversion (keeps hot loop streaming bf16) -----
__global__ __launch_bounds__(256) void cvt_weights(
    const float* __restrict__ W1, const float* __restrict__ W2,
    unsigned short* __restrict__ w1b, unsigned short* __restrict__ w2b)
{
    int i = blockIdx.x * 256 + threadIdx.x;
    if (i < FC * K1)  w1b[i] = f2bf(W1[i]);
    if (i < CO * FC)  w2b[i] = f2bf(W2[i]);
}

// --- fused main kernel ------------------------------------------------------
__global__ __launch_bounds__(256) void m2mrf_fused(
    const float* __restrict__ x,
    const unsigned short* __restrict__ w1b, const float* __restrict__ b1,
    const unsigned short* __restrict__ w2b, const float* __restrict__ b2,
    float* __restrict__ out)
{
    extern __shared__ unsigned short smem[];
    unsigned short* sCols = smem;            // [LT][K1] bf16  (128 KB)
    unsigned short* sY    = smem + LT * K1;  // [LT][FC] bf16  (128 KB)

    const int tid = threadIdx.x;
    const int blk = blockIdx.x;
    const int n   = blk >> 8;                // 256 column-blocks per image
    const int rem = blk & 255;
    const int ph  = rem >> 1;                // patch row   (0..127)
    const int pw0 = (rem & 1) * LT;          // patch col base (0 or 64)

    // ---- stage unfolded cols tile: f32 global -> bf16 LDS (coalesced) ----
    {
        const int col = tid & 63;            // patch column within tile
        const int g   = tid >> 6;            // 0..3
        const size_t xn = (size_t)n * C_IN * Hh * Ww;
        #pragma unroll 4
        for (int o = 0; o < 64; ++o) {
            const int pair = o * 4 + g;      // 0..255 = (c, kh)
            const int kh = pair & 3, c = pair >> 2;
            const float4 v = *(const float4*)(
                x + xn + ((size_t)c * Hh + (size_t)(ph * 4 + kh)) * Ww
                  + (size_t)(pw0 + col) * 4);
            const unsigned int lo = (unsigned int)f2bf(v.x) | ((unsigned int)f2bf(v.y) << 16);
            const unsigned int hi = (unsigned int)f2bf(v.z) | ((unsigned int)f2bf(v.w) << 16);
            *(uint2*)&sCols[col * K1 + (c << 4) + (kh << 2)] = make_uint2(lo, hi);
        }
    }
    __syncthreads();

    const int wave = tid >> 5;
    const int lane = tid & 31;
    const int lr   = lane & 15;              // row/col within 16
    const int lh   = lane >> 4;              // half selector

    const v8f vzero = {0.f, 0.f, 0.f, 0.f, 0.f, 0.f, 0.f, 0.f};

    // ---- GEMM1: y[1024, 64] = W1 x cols, 8 row-tiles per wave -------------
    for (int i = 0; i < 8; ++i) {
        const int rt = wave + i * 8;                       // row-tile 0..63
        v8f acc[4] = {vzero, vzero, vzero, vzero};
        const unsigned short* arow = w1b + (size_t)(rt * 16 + lr) * K1 + lh * 16;
        #pragma unroll 4
        for (int kc = 0; kc < 32; ++kc) {
            const v16bf a = *(const v16bf*)(arow + kc * 32);
            #pragma unroll
            for (int ct = 0; ct < 4; ++ct) {
                const v16bf b = *(const v16bf*)(sCols + (size_t)(ct * 16 + lr) * K1
                                                + kc * 32 + lh * 16);
                acc[ct] = __builtin_amdgcn_wmma_f32_16x16x32_bf16(
                    false, a, false, b, (short)0, acc[ct], false, false);
            }
        }
        // add b1, convert to bf16, store into sY  (y layout: [col][o])
        const int m0 = rt * 16 + lh * 8;
        const float* b1r = b1 + m0;
        #pragma unroll
        for (int ct = 0; ct < 4; ++ct) {
            unsigned int p[4];
            #pragma unroll
            for (int q = 0; q < 4; ++q) {
                const unsigned short e0 = f2bf(acc[ct][2 * q]     + b1r[2 * q]);
                const unsigned short e1 = f2bf(acc[ct][2 * q + 1] + b1r[2 * q + 1]);
                p[q] = (unsigned int)e0 | ((unsigned int)e1 << 16);
            }
            *(uint4*)&sY[(size_t)(ct * 16 + lr) * FC + m0] =
                make_uint4(p[0], p[1], p[2], p[3]);
        }
    }
    __syncthreads();

    // ---- GEMM2: z[256, 64] = W2 x y, then fold(2x2) scatter to output ----
    for (int i = 0; i < 2; ++i) {
        const int rt2 = wave + i * 8;                      // row-tile 0..15
        v8f acc[4] = {vzero, vzero, vzero, vzero};
        const unsigned short* arow = w2b + (size_t)(rt2 * 16 + lr) * FC + lh * 16;
        #pragma unroll 4
        for (int kc = 0; kc < 32; ++kc) {
            const v16bf a = *(const v16bf*)(arow + kc * 32);
            #pragma unroll
            for (int ct = 0; ct < 4; ++ct) {
                const v16bf b = *(const v16bf*)(sY + (size_t)(ct * 16 + lr) * FC
                                                + kc * 32 + lh * 16);
                acc[ct] = __builtin_amdgcn_wmma_f32_16x16x32_bf16(
                    false, a, false, b, (short)0, acc[ct], false, false);
            }
        }
        const int m0 = rt2 * 16 + lh * 8;                  // o2 base
        const float* b2r = b2 + m0;
        #pragma unroll
        for (int ct = 0; ct < 4; ++ct) {
            const int pw = pw0 + ct * 16 + lr;             // patch column
            #pragma unroll
            for (int q = 0; q < 4; ++q) {
                const int o2  = m0 + 2 * q;                // even -> kw2 = 0
                const int c   = o2 >> 2;
                const int kh2 = (o2 >> 1) & 1;
                const float2 v = make_float2(acc[ct][2 * q]     + b2r[2 * q],
                                             acc[ct][2 * q + 1] + b2r[2 * q + 1]);
                // out[n, c, ph*2 + kh2, pw*2 + {0,1}]
                *(float2*)(out + (((size_t)(n * C_IN + c) * HO)
                                  + (size_t)(ph * 2 + kh2)) * WO
                               + (size_t)pw * 2) = v;
            }
        }
    }
}

extern "C" void kernel_launch(void* const* d_in, const int* in_sizes, int n_in,
                              void* d_out, int out_size, void* d_ws, size_t ws_size,
                              hipStream_t stream) {
    const float* x  = (const float*)d_in[0];
    const float* W1 = (const float*)d_in[1];
    const float* b1 = (const float*)d_in[2];
    const float* W2 = (const float*)d_in[3];
    const float* b2 = (const float*)d_in[4];
    float* out = (float*)d_out;

    unsigned short* w1b = (unsigned short*)d_ws;           // 1024*1024 bf16
    unsigned short* w2b = w1b + (size_t)FC * K1;           //  256*1024 bf16

    // 1) convert weights to bf16 (deterministic, every call)
    cvt_weights<<<(FC * K1 + 255) / 256, 256, 0, stream>>>(W1, W2, w1b, w2b);

    // 2) fused unfold + GEMM1 + GEMM2 + fold
    //    grid: 4 images * (16384 / 64) column blocks = 1024 blocks
    //    dynamic LDS: 2 * 64 * 1024 * 2B = 256 KB (<= 320 KB per WGP)
    const size_t lds_bytes = (size_t)2 * LT * K1 * sizeof(unsigned short);
    m2mrf_fused<<<1024, 256, lds_bytes, stream>>>(x, w1b, b1, w2b, b2, out);
}
